// SchNet_16234976379045
// MI455X (gfx1250) — matrix operand order repaired and checked by reading, not compiled, not run on hardware
//
#include <hip/hip_runtime.h>
#include <hip/hip_bf16.h>

// ---- Problem constants (from reference) ----
#define BB 8
#define AA 1024
#define NNB 64
#define GG 25
#define GP 32          // G padded to WMMA K
#define FF 128
#define LL 3
#define CUTOFF_R 5.0f

typedef __bf16 bf16_t;
typedef __attribute__((ext_vector_type(16))) __bf16 v16bf;
typedef __attribute__((ext_vector_type(8)))  __bf16 v8bf;
typedef __attribute__((ext_vector_type(8)))  float  v8f;

// shifted softplus: softplus(x) - ln(2), numerically stable
__device__ __forceinline__ float sspf(float x) {
    float ax = fabsf(x);
    return fmaxf(x, 0.0f) + log1pf(__expf(-ax)) - 0.6931471805599453f;
}

__device__ __forceinline__ v8f wmma_bf16(v16bf a, v16bf b, v8f c) {
    return __builtin_amdgcn_wmma_f32_16x16x32_bf16(
        /*neg_a=*/false, a, /*neg_b=*/false, b,
        /*c_mod=*/(short)0, c, /*reuse_a=*/false, /*reuse_b=*/false);
}

// Async global->LDS copy of 16 bytes per lane (GLOBAL_LOAD_ASYNC_TO_LDS_B128,
// tracked by ASYNCcnt). LDS byte offset = low 32 bits of the generic pointer
// (aperture addressing, ISA 10.2).
__device__ __forceinline__ void async_copy16(const void* g, void* l) {
    unsigned lds_off = (unsigned)(unsigned long long)l;
    unsigned long long ga = (unsigned long long)g;
    asm volatile("global_load_async_to_lds_b128 %0, %1, off"
                 :: "v"(lds_off), "v"(ga) : "memory");
}
__device__ __forceinline__ void async_wait0() {
    asm volatile("s_wait_asynccnt 0" ::: "memory");
}

// A-fragment (16x32 bf16, ISA 7.12.2): lane m=lane&15, half=lane>>4.
// element e in [0,8):  K = kbase + 8*half + e
// element e in [8,16): K = kbase + 16 + 8*half + (e-8)
// -> two contiguous 8-element (16B) loads at kbase+8*half and +16.
__device__ __forceinline__ v16bf a_frag_bf16(const bf16_t* p0) {
    v8bf lo = *(const v8bf*)p0;
    v8bf hi = *(const v8bf*)(p0 + 16);
    v16bf a;
#pragma unroll
    for (int i = 0; i < 8; ++i) { a[i] = lo[i]; a[i + 8] = hi[i]; }
    return a;
}

// B-fragment (32x16 bf16): lane holds column N=lane&15, K = kbase + 8*half + e (e<8),
// kbase + 16 + 8*half + (e-8) (e>=8). Weights pre-transposed to [N][Ktot] -> two 16B loads.
__device__ __forceinline__ v16bf b_frag(const bf16_t* wT, int Ktot, int col,
                                        int kbase, int half) {
    return a_frag_bf16(wT + (size_t)col * Ktot + kbase + half * 8);
}

// ---------------- weight packing: f32 [L][Ksrc][F] -> bf16 [L][F][Kpad] ------
__global__ __launch_bounds__(256) void k_pack_wT(const float* __restrict__ src,
                                                 bf16_t* __restrict__ dst,
                                                 int Ksrc, int Kpad, int total) {
    int idx = blockIdx.x * 256 + threadIdx.x;
    if (idx >= total) return;
    int l = idx / (FF * Kpad);
    int r = idx - l * (FF * Kpad);
    int n = r / Kpad;
    int k = r - n * Kpad;
    float v = (k < Ksrc) ? src[((size_t)l * Ksrc + k) * FF + n] : 0.0f;
    dst[idx] = (bf16_t)v;
}

// ---------------- embedding: x[b,a,:] = emb[Z[b,a],:] (f32 + bf16 mirror) ----
__global__ __launch_bounds__(256) void k_embed(const int* __restrict__ Z,
                                               const float* __restrict__ emb,
                                               float* __restrict__ x,
                                               bf16_t* __restrict__ xbf) {
    int idx = blockIdx.x * 256 + threadIdx.x;   // over B*A*F
    int row = idx / FF, f = idx - row * FF;
    float v = emb[(size_t)Z[row] * FF + f];
    x[idx]   = v;
    xbf[idx] = (bf16_t)v;
}

// ---------------- distances r_ij ---------------------------------------------
__global__ __launch_bounds__(256) void k_dist(const float* __restrict__ pos,
                                              const float* __restrict__ cell,
                                              const float* __restrict__ coff,
                                              const int* __restrict__ nbr,
                                              float* __restrict__ r_out) {
    int idx = blockIdx.x * 256 + threadIdx.x;   // over B*A*NN
    if (idx >= BB * AA * NNB) return;
    int b   = idx / (AA * NNB);
    int rem = idx - b * (AA * NNB);
    int a   = rem / NNB;
    int j   = nbr[idx];
    const float* pi = pos + ((size_t)b * AA + a) * 3;
    const float* pj = pos + ((size_t)b * AA + j) * 3;
    const float* of = coff + (size_t)idx * 3;
    const float* cl = cell + (size_t)b * 9;
    float d2 = 0.0f;
#pragma unroll
    for (int c = 0; c < 3; ++c) {
        float dv = pj[c] - pi[c] + of[0] * cl[0 * 3 + c] + of[1] * cl[1 * 3 + c]
                 + of[2] * cl[2 * 3 + c];
        d2 += dv * dv;
    }
    r_out[idx] = (d2 > 0.0f) ? sqrtf(d2) : 0.0f;
}

// ---------------- y = xbf @ in2f (no bias), [8192x128]*[128x128] -------------
// A-tile (16x128 bf16, 4KB, contiguous) staged via async global->LDS.
__global__ __launch_bounds__(256) void k_gemm_in2f(const bf16_t* __restrict__ xbf,
                                                   const bf16_t* __restrict__ wT,
                                                   float* __restrict__ y) {
    __shared__ bf16_t xt[16][FF];
    const int tid = threadIdx.x, lane = tid & 31, wave = tid >> 5;
    const int half = lane >> 4, l15 = lane & 15;
    const int rowbase = blockIdx.x * 16;
    const int n0 = wave * 16;

    __builtin_prefetch(wT, 0, 1);
    // stage A tile: 256 lanes x 16B = 4KB
    async_copy16(xbf + (size_t)rowbase * FF + tid * 8, &xt[0][0] + tid * 8);

    v16bf bfr[4];
#pragma unroll
    for (int kk = 0; kk < 4; ++kk)
        bfr[kk] = b_frag(wT, FF, n0 + l15, kk * 32, half);

    async_wait0();
    __syncthreads();

    v8f acc = {};
#pragma unroll
    for (int kk = 0; kk < 4; ++kk) {
        v16bf a = a_frag_bf16(&xt[l15][kk * 32 + half * 8]);
        acc = wmma_bf16(a, bfr[kk], acc);
    }
#pragma unroll
    for (int k = 0; k < 8; ++k)
        y[(size_t)(rowbase + k + 8 * half) * FF + n0 + l15] = acc[k];
}

// ---------------- fused filter-net + cfconv reduction ------------------------
// One block per (b,a): 64 neighbor rows, 8 waves cover the 128 output columns.
__global__ __launch_bounds__(256) void k_cfconv(
    const float* __restrict__ r_ws, const int* __restrict__ nbr,
    const float* __restrict__ mask, const float* __restrict__ y,
    const bf16_t* __restrict__ fw1T, const float* __restrict__ fb1,
    const bf16_t* __restrict__ fw2T, const float* __restrict__ fb2,
    bf16_t* __restrict__ vout) {
    __shared__ bf16_t fij[NNB][GP];    // gaussian-smeared distances (bf16)
    __shared__ bf16_t hbuf[16][FF];    // ssp(fij@fw1+b) staging, one 16-row tile

    const int tid = threadIdx.x, lane = tid & 31, wave = tid >> 5;
    const int half = lane >> 4, l15 = lane & 15;
    const int pair = blockIdx.x;                // b*A + a
    const int b = pair >> 10;                   // A = 1024
    const int baserow = pair * NNB;
    const int n0 = wave * 16;

    __builtin_prefetch(fw1T, 0, 1);
    __builtin_prefetch(fw2T, 0, 1);

    // gaussian smearing: f_ij[n,g] = exp(coeff*(r - g*width)^2), pad g>=25 -> 0
    const float width = CUTOFF_R / (float)(GG - 1);
    const float coeff = -0.5f / (width * width);
    for (int idx = tid; idx < NNB * GP; idx += 256) {
        int row = idx >> 5, g = idx & 31;
        float rv = r_ws[baserow + row];
        float t  = rv - (float)g * width;
        float fv = (g < GG) ? __expf(coeff * t * t) : 0.0f;
        fij[row][g] = (bf16_t)fv;
    }

    // per-wave B fragments (transposed bf16 weights, L2-resident)
    v16bf bf1 = b_frag(fw1T, GP, n0 + l15, 0, half);
    v16bf bf2[4];
#pragma unroll
    for (int kk = 0; kk < 4; ++kk)
        bf2[kk] = b_frag(fw2T, FF, n0 + l15, kk * 32, half);
    const float bias1 = fb1[n0 + l15];
    const float bias2 = fb2[n0 + l15];

    float colsum = 0.0f;
    __syncthreads();

#pragma unroll
    for (int mt = 0; mt < 4; ++mt) {           // 4 tiles of 16 neighbors
        // h = ssp(fij_tile @ fw1 + fb1)  (single K=32 WMMA per wave)
        v16bf a1 = a_frag_bf16(&fij[16 * mt + l15][half * 8]);
        v8f hacc = {};
        hacc = wmma_bf16(a1, bf1, hacc);

        __syncthreads();                       // prior tile's hbuf reads done
#pragma unroll
        for (int k = 0; k < 8; ++k)
            hbuf[k + 8 * half][n0 + l15] = (bf16_t)sspf(hacc[k] + bias1);
        __syncthreads();                       // full h tile visible

        // W = h @ fw2 + fb2  (K=128 -> 4 chained WMMA)
        v8f wacc = {};
#pragma unroll
        for (int kk = 0; kk < 4; ++kk) {
            v16bf ah = a_frag_bf16(&hbuf[l15][kk * 32 + half * 8]);
            wacc = wmma_bf16(ah, bf2[kk], wacc);
        }

        // gather y[neighbor], elementwise, reduce over neighbor rows.
        // C layout: lane holds fixed column (n0+l15), rows k+8*half.
#pragma unroll
        for (int k = 0; k < 8; ++k) {
            int gr = baserow + 16 * mt + k + 8 * half;
            int nj = nbr[gr];
            float yv = y[((size_t)b * AA + nj) * FF + n0 + l15];
            colsum += (wacc[k] + bias2) * yv * mask[gr];
        }
    }
    // rows 0-7 live in lanes 0-15, rows 8-15 in lanes 16-31: combine halves
    colsum += __shfl_xor(colsum, 16, 32);
    if (lane < 16) vout[(size_t)pair * FF + n0 + l15] = (bf16_t)colsum;
}

// ---------------- post: x += ssp(v@f2out+b) @ dense + b ----------------------
// v is bf16; its 16x128 tile staged via async global->LDS.
__global__ __launch_bounds__(256) void k_post(const bf16_t* __restrict__ v,
                                              const bf16_t* __restrict__ f2outT,
                                              const float* __restrict__ f2out_b,
                                              const bf16_t* __restrict__ denseT,
                                              const float* __restrict__ dense_b,
                                              float* __restrict__ x,
                                              bf16_t* __restrict__ xbf) {
    __shared__ bf16_t vt[16][FF];
    __shared__ bf16_t tbuf[16][FF];
    const int tid = threadIdx.x, lane = tid & 31, wave = tid >> 5;
    const int half = lane >> 4, l15 = lane & 15;
    const int rowbase = blockIdx.x * 16;
    const int n0 = wave * 16;

    __builtin_prefetch(f2outT, 0, 1);
    __builtin_prefetch(denseT, 0, 1);
    async_copy16(v + (size_t)rowbase * FF + tid * 8, &vt[0][0] + tid * 8);

    v16bf bA[4], bD[4];
#pragma unroll
    for (int kk = 0; kk < 4; ++kk) {
        bA[kk] = b_frag(f2outT, FF, n0 + l15, kk * 32, half);
        bD[kk] = b_frag(denseT, FF, n0 + l15, kk * 32, half);
    }

    async_wait0();
    __syncthreads();

    // t = ssp(v @ f2out + b)
    v8f acc = {};
#pragma unroll
    for (int kk = 0; kk < 4; ++kk) {
        v16bf a = a_frag_bf16(&vt[l15][kk * 32 + half * 8]);
        acc = wmma_bf16(a, bA[kk], acc);
    }
    float b1 = f2out_b[n0 + l15];
#pragma unroll
    for (int k = 0; k < 8; ++k)
        tbuf[k + 8 * half][n0 + l15] = (bf16_t)sspf(acc[k] + b1);
    __syncthreads();

    // x += t @ dense + b  (update f32 master and bf16 mirror)
    v8f acc2 = {};
#pragma unroll
    for (int kk = 0; kk < 4; ++kk) {
        v16bf a = a_frag_bf16(&tbuf[l15][kk * 32 + half * 8]);
        acc2 = wmma_bf16(a, bD[kk], acc2);
    }
    float b2 = dense_b[n0 + l15];
#pragma unroll
    for (int k = 0; k < 8; ++k) {
        size_t o = (size_t)(rowbase + k + 8 * half) * FF + n0 + l15;
        float nx = x[o] + acc2[k] + b2;
        x[o]   = nx;
        xbf[o] = (bf16_t)nx;
    }
}

extern "C" void kernel_launch(void* const* d_in, const int* in_sizes, int n_in,
                              void* d_out, int out_size, void* d_ws, size_t ws_size,
                              hipStream_t stream) {
    const int*   Z        = (const int*)d_in[0];
    const float* pos      = (const float*)d_in[1];
    const float* cell     = (const float*)d_in[2];
    const float* coff     = (const float*)d_in[3];
    const int*   nbr      = (const int*)d_in[4];
    const float* nmask    = (const float*)d_in[5];
    const float* emb      = (const float*)d_in[6];
    const float* fw1      = (const float*)d_in[7];
    const float* fb1      = (const float*)d_in[8];
    const float* fw2      = (const float*)d_in[9];
    const float* fb2      = (const float*)d_in[10];
    const float* in2f_w   = (const float*)d_in[11];
    const float* f2out_w  = (const float*)d_in[12];
    const float* f2out_b  = (const float*)d_in[13];
    const float* dense_w  = (const float*)d_in[14];
    const float* dense_b  = (const float*)d_in[15];
    float* x = (float*)d_out;                       // live activation buffer

    // workspace carve-up
    char* wp = (char*)d_ws;
    float*  r_ws = (float*)wp;  wp += (size_t)BB * AA * NNB * 4;   // 2 MB
    float*  ybuf = (float*)wp;  wp += (size_t)BB * AA * FF * 4;    // 4 MB
    bf16_t* xbf  = (bf16_t*)wp; wp += (size_t)BB * AA * FF * 2;    // 2 MB
    bf16_t* vbuf = (bf16_t*)wp; wp += (size_t)BB * AA * FF * 2;    // 2 MB
    bf16_t* fw1T   = (bf16_t*)wp; wp += (size_t)LL * FF * GP * 2;
    bf16_t* fw2T   = (bf16_t*)wp; wp += (size_t)LL * FF * FF * 2;
    bf16_t* in2fT  = (bf16_t*)wp; wp += (size_t)LL * FF * FF * 2;
    bf16_t* f2outT = (bf16_t*)wp; wp += (size_t)LL * FF * FF * 2;
    bf16_t* denseT = (bf16_t*)wp; wp += (size_t)LL * FF * FF * 2;

    // pack weights -> transposed bf16 (fw1 K-padded 25->32)
    {
        int t1 = LL * FF * GP;
        int t2 = LL * FF * FF;
        k_pack_wT<<<(t1 + 255) / 256, 256, 0, stream>>>(fw1, fw1T, GG, GP, t1);
        k_pack_wT<<<(t2 + 255) / 256, 256, 0, stream>>>(fw2, fw2T, FF, FF, t2);
        k_pack_wT<<<(t2 + 255) / 256, 256, 0, stream>>>(in2f_w, in2fT, FF, FF, t2);
        k_pack_wT<<<(t2 + 255) / 256, 256, 0, stream>>>(f2out_w, f2outT, FF, FF, t2);
        k_pack_wT<<<(t2 + 255) / 256, 256, 0, stream>>>(dense_w, denseT, FF, FF, t2);
    }

    k_embed<<<(BB * AA * FF) / 256, 256, 0, stream>>>(Z, emb, x, xbf);
    k_dist<<<(BB * AA * NNB + 255) / 256, 256, 0, stream>>>(pos, cell, coff, nbr, r_ws);

    for (int l = 0; l < LL; ++l) {
        k_gemm_in2f<<<BB * AA / 16, 256, 0, stream>>>(xbf, in2fT + (size_t)l * FF * FF, ybuf);
        k_cfconv<<<BB * AA, 256, 0, stream>>>(r_ws, nbr, nmask, ybuf,
                                              fw1T + (size_t)l * FF * GP, fb1 + l * FF,
                                              fw2T + (size_t)l * FF * FF, fb2 + l * FF,
                                              vbuf);
        k_post<<<BB * AA / 16, 256, 0, stream>>>(vbuf,
                                                 f2outT + (size_t)l * FF * FF, f2out_b + l * FF,
                                                 denseT + (size_t)l * FF * FF, dense_b + l * FF,
                                                 x, xbf);
    }
}
